// DiseaseGraphGAT_81896436400225
// MI455X (gfx1250) — matrix operand order repaired and checked by reading, not compiled, not run on hardware
//
#include <hip/hip_runtime.h>

typedef __attribute__((ext_vector_type(16))) _Float16 v16h;
typedef __attribute__((ext_vector_type(8)))  float    v8f;
typedef __attribute__((ext_vector_type(4)))  float    f32x4;
typedef __attribute__((ext_vector_type(4)))  int      i32x4;

#define N 8192
#define D 128

// ---- monotonic float<->uint encoding for atomicMax over signed floats ----
__device__ __forceinline__ unsigned enc_f32(float f) {
  unsigned u = __float_as_uint(f);
  return (u & 0x80000000u) ? ~u : (u | 0x80000000u);
}
__device__ __forceinline__ float dec_f32(unsigned u) {
  unsigned b = (u & 0x80000000u) ? (u ^ 0x80000000u) : ~u;
  return __uint_as_float(b);
}

__global__ void k_init(unsigned* smax_u) { *smax_u = 0u; }

// ---- s_i[n] = emb[n]·attn[:D], s_j[n] = emb[n]·attn[D:], smax = max s_j ----
__global__ __launch_bounds__(256) void k_prep_s(
    const float* __restrict__ emb, const float* __restrict__ attn,
    float* __restrict__ s_i, float* __restrict__ s_j, unsigned* __restrict__ smax_u)
{
  const int lane = threadIdx.x & 31;
  const int wave = threadIdx.x >> 5;
  const int row  = blockIdx.x * 8 + wave;
  const f32x4* e4 = (const f32x4*)(emb + (size_t)row * D);
  const f32x4* a4 = (const f32x4*)attn;
  f32x4 e  = e4[lane];          // 4 floats per lane, 32 lanes = 128
  f32x4 a0 = a4[lane];          // attn[:128]
  f32x4 a1 = a4[32 + lane];     // attn[128:256]
  float si = e.x*a0.x + e.y*a0.y + e.z*a0.z + e.w*a0.w;
  float sj = e.x*a1.x + e.y*a1.y + e.z*a1.z + e.w*a1.w;
  #pragma unroll
  for (int off = 16; off >= 1; off >>= 1) {
    si += __shfl_xor(si, off, 32);
    sj += __shfl_xor(sj, off, 32);
  }
  if (lane == 0) {
    s_i[row] = si;
    s_j[row] = sj;
    atomicMax(smax_u, enc_f32(sj));
  }
}

// ---- embT[d][n] = (f16) emb[n][d] : LDS-tiled transpose, 64 rows/block ----
__global__ __launch_bounds__(256) void k_prep_t(
    const float* __restrict__ emb, _Float16* __restrict__ embT)
{
  __shared__ float tile[64 * 129];   // +1 pad: avoid 64-way bank conflicts
  const int n0  = blockIdx.x * 64;
  const int tid = threadIdx.x;
  #pragma unroll
  for (int it = 0; it < 32; ++it) {
    int idx = it * 256 + tid;
    int nl = idx >> 7;       // 0..63
    int d  = idx & 127;
    tile[nl * 129 + d] = emb[(size_t)(n0 + nl) * D + d];   // coalesced read
  }
  __syncthreads();
  #pragma unroll
  for (int it = 0; it < 32; ++it) {
    int idx = it * 256 + tid;
    int d  = idx >> 6;       // 0..127
    int nl = idx & 63;
    embT[(size_t)d * N + n0 + nl] = (_Float16)tile[nl * 129 + d];
  }
}

// ---- fused masked-softmax-GEMM: 16 rows per block, 8 waves split columns ----
__global__ __launch_bounds__(256) void k_gat_main(
    const int* __restrict__ adj, const _Float16* __restrict__ embT,
    const float* __restrict__ s_i, const float* __restrict__ s_j,
    const unsigned* __restrict__ smax_u, float* __restrict__ out)
{
  __shared__ float lds_acc[16 * D];
  __shared__ float lds_sum[16];
  const int tid  = threadIdx.x;
  const int lane = tid & 31;
  // wave id is wave-uniform: force it into an SGPR so the j-loop is scalar
  const int wave = __builtin_amdgcn_readfirstlane(tid >> 5);
  const int m    = lane & 15;     // A row / C column within 16
  const int hi   = lane >> 4;     // lane half
  const int i0   = blockIdx.x * 16;

  for (int idx = tid; idx < 16 * D; idx += 256) lds_acc[idx] = 0.0f;
  if (tid < 16) lds_sum[tid] = 0.0f;
  __syncthreads();

  const float smax = dec_f32(*smax_u);
  const float si   = s_i[i0 + m];
  const float ubx  = si + smax;
  const float ub   = fmaxf(ubx, 0.2f * ubx);   // fixed softmax shift >= any row score
  const float L2E  = 1.44269504088896340736f;

  v8f acc[8] = {};        // 8 D-tiles of the 16x16 f32 C layout
  v8f accs   = {};        // row-sum accumulator (ones-B trick)
  v16h ones;
  #pragma unroll
  for (int h = 0; h < 16; ++h) ones[h] = (_Float16)1.0f;

  const int jbeg  = wave * (N / 8);            // scalar
  // k order used here for A and B (contiguous): lane<16 -> K = 0..15 ; lane>=16 -> K = 16..31
  const int koff1 = hi ? 16 : 0;
  const int koff2 = hi ? 24 : 8;
  const int boff  = hi ? 16 : 0;
  const int* __restrict__ arow = adj + (size_t)(i0 + m) * N;

  // ---- software pipeline: preload first tile's adj + s_j ----
  i32x4 a0 = __builtin_nontemporal_load((const i32x4*)(arow + jbeg + koff1));
  i32x4 a1 = __builtin_nontemporal_load((const i32x4*)(arow + jbeg + koff1 + 4));
  i32x4 a2 = __builtin_nontemporal_load((const i32x4*)(arow + jbeg + koff2));
  i32x4 a3 = __builtin_nontemporal_load((const i32x4*)(arow + jbeg + koff2 + 4));
  f32x4 s0 = *(const f32x4*)(s_j + jbeg + koff1);
  f32x4 s1 = *(const f32x4*)(s_j + jbeg + koff1 + 4);
  f32x4 s2 = *(const f32x4*)(s_j + jbeg + koff2);
  f32x4 s3 = *(const f32x4*)(s_j + jbeg + koff2 + 4);

  #pragma unroll 1
  for (int it = 0; it < 32; ++it) {
    const int j0 = jbeg + it * 32;

    // 1) issue all 8 B-fragment loads up front (16 x b128, single clause)
    v16h Bf[8];
    #pragma unroll
    for (int t = 0; t < 8; ++t)
      Bf[t] = *(const v16h*)(embT + (size_t)(t * 16 + m) * N + j0 + boff);

    // 2) issue next tile's adj/s_j loads (clamped uniform index, no branch)
    const int jn = (it < 31) ? (j0 + 32) : jbeg;
    i32x4 na0 = __builtin_nontemporal_load((const i32x4*)(arow + jn + koff1));
    i32x4 na1 = __builtin_nontemporal_load((const i32x4*)(arow + jn + koff1 + 4));
    i32x4 na2 = __builtin_nontemporal_load((const i32x4*)(arow + jn + koff2));
    i32x4 na3 = __builtin_nontemporal_load((const i32x4*)(arow + jn + koff2 + 4));
    f32x4 ns0 = *(const f32x4*)(s_j + jn + koff1);
    f32x4 ns1 = *(const f32x4*)(s_j + jn + koff1 + 4);
    f32x4 ns2 = *(const f32x4*)(s_j + jn + koff2);
    f32x4 ns3 = *(const f32x4*)(s_j + jn + koff2 + 4);
    __builtin_prefetch(arow + j0 + 1024, 0, 1);   // stream adj further ahead

    // 3) compute A fragment from current tile regs (overlaps in-flight loads)
    float sjv[16]; int av[16];
    *(f32x4*)(sjv + 0)  = s0; *(f32x4*)(sjv + 4)  = s1;
    *(f32x4*)(sjv + 8)  = s2; *(f32x4*)(sjv + 12) = s3;
    *(i32x4*)(av + 0)   = a0; *(i32x4*)(av + 4)   = a1;
    *(i32x4*)(av + 8)   = a2; *(i32x4*)(av + 12)  = a3;
    v16h A;
    #pragma unroll
    for (int h = 0; h < 16; ++h) {
      float x = si + sjv[h];
      float e = fmaxf(x, 0.2f * x);              // leaky_relu, branchless
      float p = exp2f((e - ub) * L2E);
      A[h] = (_Float16)((av[h] != 0) ? p : 0.0f); // cndmask, no branch
    }

    // 4) matrix ops: row-sum (ones-B) + 8 D-tiles
    accs = __builtin_amdgcn_wmma_f32_16x16x32_f16(
        false, A, false, ones, (short)0, accs, false, false);
    #pragma unroll
    for (int t = 0; t < 8; ++t)
      acc[t] = __builtin_amdgcn_wmma_f32_16x16x32_f16(
          false, A, false, Bf[t], (short)0, acc[t], false, false);

    // 5) rotate pipeline registers
    a0 = na0; a1 = na1; a2 = na2; a3 = na3;
    s0 = ns0; s1 = ns1; s2 = ns2; s3 = ns3;
  }

  // combine waves: C layout is VGPR v -> row (v + 8*hi), col m (within D-tile t)
  #pragma unroll
  for (int t = 0; t < 8; ++t)
    #pragma unroll
    for (int v = 0; v < 8; ++v)
      atomicAdd(&lds_acc[(v + 8 * hi) * D + t * 16 + m], acc[t][v]);
  if (m == 0) {
    #pragma unroll
    for (int v = 0; v < 8; ++v) atomicAdd(&lds_sum[v + 8 * hi], accs[v]);
  }
  __syncthreads();

  for (int idx = tid; idx < 16 * D; idx += 256) {
    int row = idx >> 7;
    int d   = idx & 127;
    out[(size_t)(i0 + row) * D + d] = lds_acc[idx] / lds_sum[row];
  }
}

extern "C" void kernel_launch(void* const* d_in, const int* in_sizes, int n_in,
                              void* d_out, int out_size, void* d_ws, size_t ws_size,
                              hipStream_t stream) {
  (void)in_sizes; (void)n_in; (void)out_size; (void)ws_size;
  const int*   adj  = (const int*)d_in[0];
  const float* emb  = (const float*)d_in[1];
  const float* attn = (const float*)d_in[2];
  float* out = (float*)d_out;

  char* ws = (char*)d_ws;
  unsigned*  smax_u = (unsigned*)ws;                                // 64 B slot
  float*     s_i    = (float*)(ws + 64);                            // 32 KB
  float*     s_j    = (float*)(ws + 64 + N * sizeof(float));        // 32 KB
  _Float16*  embT   = (_Float16*)(ws + 64 + 2 * N * sizeof(float)); // 2 MB

  k_init    <<<1,      1,   0, stream>>>(smax_u);
  k_prep_s  <<<N / 8,  256, 0, stream>>>(emb, attn, s_i, s_j, smax_u);
  k_prep_t  <<<N / 64, 256, 0, stream>>>(emb, embT);
  k_gat_main<<<N / 16, 256, 0, stream>>>(adj, embT, s_i, s_j, smax_u, out);
}
